// VanillaRNN_76836964925694
// MI455X (gfx1250) — matrix operand order repaired and checked by reading, not compiled, not run on hardware
//
#include <hip/hip_runtime.h>
#include <math.h>

typedef __bf16 bf16;
typedef __attribute__((ext_vector_type(16))) __bf16 v16bf;
typedef __attribute__((ext_vector_type(8)))  float  v8f;
typedef __attribute__((ext_vector_type(4)))  unsigned int u32x4;

union Frag16 {           // 32 bytes = 16 bf16 = one WMMA A/B fragment per lane
    u32x4 q[2];
    v16bf v;
};

// Fast tanh: 1 - 2/(e^{2x}+1). v_exp_f32 + v_rcp_f32; exact at +/-1 saturation.
__device__ __forceinline__ float fast_tanh(float x) {
    float e = __expf(2.0f * x);                     // v_exp_f32 (TRANS)
    float r = __builtin_amdgcn_rcpf(e + 1.0f);      // v_rcp_f32 (TRANS)
    return fmaf(-2.0f, r, 1.0f);
}

// ---------- one-time: fp32 [R x C] -> bf16 transpose [C x R] ----------
__global__ void k_transpose_bf16(const float* __restrict__ W, bf16* __restrict__ Wt,
                                 int R, int C) {
    int idx = blockIdx.x * blockDim.x + threadIdx.x;
    if (idx >= R * C) return;
    int c = idx / R;
    int r = idx - c * R;
    Wt[idx] = (bf16)W[(size_t)r * C + c];
}

// ---------- one-time: xhb = x @ W_hx + bias_h ; h1 = tanh(xhb) ----------
__global__ void k_input_proj(const float* __restrict__ x, const float* __restrict__ W_hx,
                             const float* __restrict__ bias_h,
                             float* __restrict__ xhb, bf16* __restrict__ h0,
                             int Din, int H) {
    int h = blockIdx.x * blockDim.x + threadIdx.x;
    int b = blockIdx.y;
    if (h >= H) return;
    float acc = bias_h[h];
    const float* xr = x + (size_t)b * Din;
    #pragma unroll 4
    for (int k = 0; k < Din; ++k)
        acc = fmaf(xr[k], W_hx[(size_t)k * H + h], acc);
    xhb[(size_t)b * H + h] = acc;
    h0[(size_t)b * H + h] = (bf16)fast_tanh(acc);   // first recurrence step (h_prev = 0)
}

// ---------- hot kernel: D = A[512xK](bf16) x Bt[NxK]^T (bf16), fp32 acc ----------
// Wave tile 32(M) x 64(N): 2 A-fragments, 4 B-fragments, 8 WMMA per K-step.
// Block = 2 waves stacked along M -> 64x64 block tile.
// TANH path (recurrence): out_bf16 = tanh(D + xhb)
// head path:              out_f32  = D + bias_p
template<bool TANH>
__global__ __launch_bounds__(64)
void k_gemm_bf16(const bf16* __restrict__ A,     // [M x K] row-major (hidden state)
                 const bf16* __restrict__ Bt,    // [N x K] row-major (pre-transposed weights)
                 const float* __restrict__ Cadd, // [M x N] xhb (TANH) or nullptr
                 const float* __restrict__ biasN,// [N] (head) or nullptr
                 bf16* __restrict__ outBf,
                 float* __restrict__ outF,
                 int N, int K) {
    const int lane = threadIdx.x & 31;
    const int wave = threadIdx.x >> 5;
    const int half = lane >> 4;     // 0: lanes 0-15, 1: lanes 16-31
    const int r16  = lane & 15;

    const int m0 = blockIdx.x * 64 + wave * 32;  // 2 waves stacked along M
    const int n0 = blockIdx.y * 64;              // each wave covers 64 cols

    // A 16x32 bf16 fragment: lane half 0 holds row r16, K {0..15};
    // half 1 holds K {16..31}  -> one contiguous 32B chunk.
    const bf16* ap0 = A + (size_t)(m0 + r16) * K + half * 16;
    const bf16* ap1 = ap0 + (size_t)16 * K;
    // B 32x16 bf16 fragment: lane holds column (n0+j*16+r16),
    // half 0 -> K {0..15}, half 1 -> K {16..31} -> one contiguous 32B chunk.
    const bf16* bp0 = Bt + (size_t)(n0 +  0 + r16) * K + half * 16;
    const bf16* bp1 = Bt + (size_t)(n0 + 16 + r16) * K + half * 16;
    const bf16* bp2 = Bt + (size_t)(n0 + 32 + r16) * K + half * 16;
    const bf16* bp3 = Bt + (size_t)(n0 + 48 + r16) * K + half * 16;

    v8f acc00 = {}, acc01 = {}, acc02 = {}, acc03 = {};
    v8f acc10 = {}, acc11 = {}, acc12 = {}, acc13 = {};

    #pragma unroll 2
    for (int k0 = 0; k0 < K; k0 += 32) {
        Frag16 a0, a1, b0, b1, b2, b3;
        a0.q[0] = *(const u32x4*)(ap0 + k0);
        a0.q[1] = *(const u32x4*)(ap0 + k0 + 8);
        a1.q[0] = *(const u32x4*)(ap1 + k0);
        a1.q[1] = *(const u32x4*)(ap1 + k0 + 8);
        b0.q[0] = *(const u32x4*)(bp0 + k0);
        b0.q[1] = *(const u32x4*)(bp0 + k0 + 8);
        b1.q[0] = *(const u32x4*)(bp1 + k0);
        b1.q[1] = *(const u32x4*)(bp1 + k0 + 8);
        b2.q[0] = *(const u32x4*)(bp2 + k0);
        b2.q[1] = *(const u32x4*)(bp2 + k0 + 8);
        b3.q[0] = *(const u32x4*)(bp3 + k0);
        b3.q[1] = *(const u32x4*)(bp3 + k0 + 8);

        // 8 args: (neg_a, A, neg_b, B, c_mod, C, reuse_a, reuse_b)
        acc00 = __builtin_amdgcn_wmma_f32_16x16x32_bf16(false, a0.v, false, b0.v, (short)0, acc00, false, false);
        acc01 = __builtin_amdgcn_wmma_f32_16x16x32_bf16(false, a0.v, false, b1.v, (short)0, acc01, false, false);
        acc02 = __builtin_amdgcn_wmma_f32_16x16x32_bf16(false, a0.v, false, b2.v, (short)0, acc02, false, false);
        acc03 = __builtin_amdgcn_wmma_f32_16x16x32_bf16(false, a0.v, false, b3.v, (short)0, acc03, false, false);
        acc10 = __builtin_amdgcn_wmma_f32_16x16x32_bf16(false, a1.v, false, b0.v, (short)0, acc10, false, false);
        acc11 = __builtin_amdgcn_wmma_f32_16x16x32_bf16(false, a1.v, false, b1.v, (short)0, acc11, false, false);
        acc12 = __builtin_amdgcn_wmma_f32_16x16x32_bf16(false, a1.v, false, b2.v, (short)0, acc12, false, false);
        acc13 = __builtin_amdgcn_wmma_f32_16x16x32_bf16(false, a1.v, false, b3.v, (short)0, acc13, false, false);
    }

    // C/D layout: row = m0 + i*16 + half*8 + vgpr_r, col = n0 + j*16 + (lane&15).
    v8f accs[2][4] = {{acc00, acc01, acc02, acc03}, {acc10, acc11, acc12, acc13}};
    #pragma unroll
    for (int i = 0; i < 2; ++i) {
        const int mbase = m0 + i * 16 + half * 8;
        #pragma unroll
        for (int j = 0; j < 4; ++j) {
            const int n = n0 + j * 16 + r16;
            #pragma unroll
            for (int r = 0; r < 8; ++r) {
                const size_t idx = (size_t)(mbase + r) * N + n;
                float v = accs[i][j][r];
                if (TANH) {
                    outBf[idx] = (bf16)fast_tanh(v + Cadd[idx]);
                } else {
                    outF[idx] = v + biasN[n];
                }
            }
        }
    }
}

// ---------- softmax over classes, transposed output y[c, b] ----------
__global__ void k_softmax_t(const float* __restrict__ p, float* __restrict__ y,
                            int B, int C) {
    const int b = blockIdx.x;
    const int c = threadIdx.x;            // C == 128 threads
    __shared__ float sm[128];
    float v = p[(size_t)b * C + c];
    sm[c] = v;
    __syncthreads();
    for (int s = 64; s > 0; s >>= 1) {
        if (c < s) sm[c] = fmaxf(sm[c], sm[c + s]);
        __syncthreads();
    }
    float mx = sm[0];
    __syncthreads();
    float e = expf(v - mx);
    sm[c] = e;
    __syncthreads();
    for (int s = 64; s > 0; s >>= 1) {
        if (c < s) sm[c] += sm[c + s];
        __syncthreads();
    }
    y[(size_t)c * B + b] = e / sm[0];
}

extern "C" void kernel_launch(void* const* d_in, const int* in_sizes, int n_in,
                              void* d_out, int out_size, void* d_ws, size_t ws_size,
                              hipStream_t stream) {
    (void)in_sizes; (void)n_in; (void)out_size; (void)ws_size;
    const float* x      = (const float*)d_in[0];
    const float* W_hx   = (const float*)d_in[1];
    const float* W_hh   = (const float*)d_in[2];
    const float* bias_h = (const float*)d_in[3];
    const float* W_ph   = (const float*)d_in[4];
    const float* bias_p = (const float*)d_in[5];
    // d_in[6] = seq_length (device scalar); fixed at 128 per problem spec —
    // cannot be read synchronously under graph capture.

    constexpr int B = 512, DIN = 256, H = 768, C = 128, T = 128;

    char* ws = (char*)d_ws;
    size_t off = 0;
    auto carve = [&](size_t bytes) -> void* {
        off = (off + 255) & ~(size_t)255;
        void* p = ws + off;
        off += bytes;
        return p;
    };
    bf16*  W_hhT = (bf16*) carve((size_t)H * H * sizeof(bf16)); // [H x H] = W_hh^T
    bf16*  W_phT = (bf16*) carve((size_t)C * H * sizeof(bf16)); // [C x H] = W_ph^T
    float* xhb   = (float*)carve((size_t)B * H * sizeof(float));
    bf16*  hbuf0 = (bf16*) carve((size_t)B * H * sizeof(bf16));
    bf16*  hbuf1 = (bf16*) carve((size_t)B * H * sizeof(bf16));
    float* pbuf  = (float*)carve((size_t)B * C * sizeof(float));

    k_transpose_bf16<<<dim3((H * H + 255) / 256), dim3(256), 0, stream>>>(W_hh, W_hhT, H, H);
    k_transpose_bf16<<<dim3((H * C + 255) / 256), dim3(256), 0, stream>>>(W_ph, W_phT, H, C);
    k_input_proj<<<dim3(H / 256, B), dim3(256), 0, stream>>>(x, W_hx, bias_h, xhb, hbuf0, DIN, H);

    // steps 2..T (step 1 folded into k_input_proj)
    bf16* hc = hbuf0;
    bf16* hn = hbuf1;
    for (int t = 1; t < T; ++t) {
        k_gemm_bf16<true><<<dim3(B / 64, H / 64), dim3(64), 0, stream>>>(
            hc, W_hhT, xhb, nullptr, hn, nullptr, H, H);
        bf16* tmp = hc; hc = hn; hn = tmp;
    }

    // head: p = h_last @ W_ph + bias_p
    k_gemm_bf16<false><<<dim3(B / 64, C / 64), dim3(64), 0, stream>>>(
        hc, W_phT, nullptr, bias_p, nullptr, pbuf, C, H);

    k_softmax_t<<<dim3(B), dim3(C), 0, stream>>>(pbuf, (float*)d_out, B, C);
}
